// AttentionLayer_18107582120164
// MI455X (gfx1250) — matrix operand order repaired and checked by reading, not compiled, not run on hardware
//
#include <hip/hip_runtime.h>

// ---------------------------------------------------------------------------
// Shapes (compile-time constants from the reference)
// ---------------------------------------------------------------------------
#define B_   2
#define T_   2048
#define D_   1024
#define H_   16
#define HD_  64
#define M_   (B_ * T_)      // 4096 rows for the projections

typedef __attribute__((ext_vector_type(16))) __bf16 v16bf;
typedef __attribute__((ext_vector_type(8)))  float  v8f;

union FragBF {                // one WMMA A/B operand (16 bf16 per lane = 32B)
    v16bf v;
    uint4 q[2];
};

__device__ __forceinline__ unsigned short f2bf(float f) {
    // round-to-nearest-even f32 -> bf16
    unsigned int u = __float_as_uint(f);
    unsigned int r = (u + 0x7FFFu + ((u >> 16) & 1u)) >> 16;
    return (unsigned short)r;
}

// ---------------------------------------------------------------------------
// CDNA5 async global->LDS copy (ASYNCcnt path, no VGPR data round-trip).
// dst operand = VGPR holding LDS byte offset (low 32 bits of generic pointer);
// addr operand = 64-bit global address.  ISA 15.18.3 opcode 98.
// ---------------------------------------------------------------------------
__device__ __forceinline__ void async_ld16(void* lds, const void* g) {
    unsigned lds_off = (unsigned)(unsigned long long)(uintptr_t)lds;
    unsigned long long ga = (unsigned long long)(uintptr_t)g;
    asm volatile("global_load_async_to_lds_b128 %0, %1, off"
                 :: "v"(lds_off), "v"(ga)
                 : "memory");
}
__device__ __forceinline__ void async_wait0() {
    asm volatile("s_wait_asynccnt 0x0" ::: "memory");
}

// ---------------------------------------------------------------------------
// Kernel 1: f32 -> bf16 pack (8 elements / thread, 16B vector load+store)
// ---------------------------------------------------------------------------
__global__ __launch_bounds__(256) void cvt_f32_bf16(const float* __restrict__ src,
                                                    unsigned short* __restrict__ dst,
                                                    int n8) {
    int i = blockIdx.x * blockDim.x + threadIdx.x;
    if (i >= n8) return;
    const float4* s = reinterpret_cast<const float4*>(src) + (size_t)i * 2;
    float4 a = s[0], b = s[1];
    uint4 o;
    o.x = (unsigned)f2bf(a.x) | ((unsigned)f2bf(a.y) << 16);
    o.y = (unsigned)f2bf(a.z) | ((unsigned)f2bf(a.w) << 16);
    o.z = (unsigned)f2bf(b.x) | ((unsigned)f2bf(b.y) << 16);
    o.w = (unsigned)f2bf(b.z) | ((unsigned)f2bf(b.w) << 16);
    reinterpret_cast<uint4*>(dst)[i] = o;
}

// ---------------------------------------------------------------------------
// Kernel 2: projection GEMM  y = x @ W^T + bias, output bf16 row-major.
// Block tile 256(M) x 64(N), K staged 64 deep; 8 waves, wave tile 32x64
// (8 f32 accumulators, 16 WMMAs per stage, A-frag reuse x4, B-frag reuse x2).
// W row-major is already the contiguous-K layout of the B=W^T fragment.
// ---------------------------------------------------------------------------
#define GM   256
#define GN   64
#define GK   64
#define AP   72   // 64 + 8 bf16 row padding -> conflict-free 16B ds accesses

__global__ __launch_bounds__(256) void proj_gemm(const unsigned short* __restrict__ xb,
                                                 const unsigned short* __restrict__ wb,
                                                 const float* __restrict__ bias,
                                                 unsigned short* __restrict__ out) {
    __shared__ __align__(16) unsigned short At[GM * AP];
    __shared__ __align__(16) unsigned short Bt[GN * AP];

    const int tid  = threadIdx.x;
    const int w    = tid >> 5;          // wave id (wave32)
    const int l    = tid & 31;
    const int half = l >> 4;            // hi/lo 16-lane group
    const int lc   = l & 15;
    const int m0   = blockIdx.y * GM;
    const int n0   = blockIdx.x * GN;

    const v8f vzero = {0.f, 0.f, 0.f, 0.f, 0.f, 0.f, 0.f, 0.f};
    v8f acc[2][4];
#pragma unroll
    for (int mt = 0; mt < 2; ++mt)
#pragma unroll
        for (int nt = 0; nt < 4; ++nt) acc[mt][nt] = vzero;

    for (int k0 = 0; k0 < D_; k0 += GK) {
        __syncthreads();
        // --- stage A tile 256x64 bf16 (2048 chunks of 16B, 8/thread, async) ---
#pragma unroll
        for (int i = 0; i < 8; ++i) {
            int c = tid + i * 256;
            int r = c >> 3, c8 = (c & 7) * 8;
            async_ld16(At + r * AP + c8, xb + (size_t)(m0 + r) * D_ + k0 + c8);
        }
        // --- stage B tile 64x64 bf16 (512 chunks, 2/thread, async) ---
#pragma unroll
        for (int i = 0; i < 2; ++i) {
            int c = tid + i * 256;
            int r = c >> 3, c8 = (c & 7) * 8;
            async_ld16(Bt + r * AP + c8, wb + (size_t)(n0 + r) * D_ + k0 + c8);
        }
        if (k0 + GK < D_) {  // L2 prefetch of next K stage
            __builtin_prefetch(xb + (size_t)(m0 + tid) * D_ + k0 + GK, 0, 1);
            if (tid < GN)
                __builtin_prefetch(wb + (size_t)(n0 + tid) * D_ + k0 + GK, 0, 1);
        }
        async_wait0();
        __syncthreads();

#pragma unroll
        for (int kk = 0; kk < 2; ++kk) {
            // A fragments (16x32): lanes 0-15 K{0-7,16-23}, lanes 16-31 K{8-15,24-31}
            FragBF a[2];
#pragma unroll
            for (int mt = 0; mt < 2; ++mt) {
                const unsigned short* ap =
                    At + (w * 32 + mt * 16 + lc) * AP + kk * 32 + half * 8;
                a[mt].q[0] = *reinterpret_cast<const uint4*>(ap);
                a[mt].q[1] = *reinterpret_cast<const uint4*>(ap + 16);
            }
#pragma unroll
            for (int nt = 0; nt < 4; ++nt) {
                // B fragment (32x16): lane group = N, lanes 0-15 K{0-7,16-23}, lanes 16-31 K{8-15,24-31}
                FragBF bf;
                const unsigned short* bp =
                    Bt + (nt * 16 + lc) * AP + kk * 32 + half * 8;
                bf.q[0] = *reinterpret_cast<const uint4*>(bp);
                bf.q[1] = *reinterpret_cast<const uint4*>(bp + 16);
#pragma unroll
                for (int mt = 0; mt < 2; ++mt) {
                    acc[mt][nt] = __builtin_amdgcn_wmma_f32_16x16x32_bf16(
                        false, a[mt].v, false, bf.v, (short)0, acc[mt][nt],
                        false, false);
                }
            }
        }
    }

    // epilogue: + bias, store bf16 (C layout: vgpr j -> row half*8+j, col lc)
#pragma unroll
    for (int nt = 0; nt < 4; ++nt) {
        float bv = bias[n0 + nt * 16 + lc];
#pragma unroll
        for (int mt = 0; mt < 2; ++mt) {
#pragma unroll
            for (int j = 0; j < 8; ++j) {
                int row = m0 + w * 32 + mt * 16 + half * 8 + j;
                out[(size_t)row * D_ + n0 + nt * 16 + lc] = f2bf(acc[mt][nt][j] + bv);
            }
        }
    }
}

// ---------------------------------------------------------------------------
// Kernel 3: flash attention per (b,h); V = Q (reference quirk).
//   grid.y = b*16+h, grid.x = query block of 128; wave owns 16 queries.
//   Per 64-key block: S=Q K^T (8 wmma), online softmax, O += P V (8 wmma).
// ---------------------------------------------------------------------------
#define BK   64
#define KP   72    // 64 + 8 bf16 padding
#define NW   8

__global__ __launch_bounds__(256) void attn_kernel(const unsigned short* __restrict__ Qb,
                                                   const unsigned short* __restrict__ Kb,
                                                   float* __restrict__ out) {
    __shared__ __align__(16) unsigned short Kt[BK * KP];        // K rows  [t][d]
    __shared__ __align__(16) unsigned short Vt[HD_ * KP];       // V^T     [d][t]
    __shared__ __align__(16) unsigned short Pb[NW * 16 * KP];   // P per wave [q][t]

    const int tid  = threadIdx.x;
    const int w    = tid >> 5;
    const int l    = tid & 31;
    const int half = l >> 4;
    const int lc   = l & 15;
    const int bh   = blockIdx.y;
    const int b    = bh >> 4;
    const int h    = bh & 15;
    const int q0   = blockIdx.x * (NW * 16) + w * 16;
    const size_t rowbase = (size_t)b * T_ * D_;
    const int hc   = h * HD_;

    // Q fragments for this wave's 16 rows, both 32-deep k-steps of HD=64
    FragBF qf[2];
#pragma unroll
    for (int kk = 0; kk < 2; ++kk) {
        const unsigned short* qp =
            Qb + rowbase + (size_t)(q0 + lc) * D_ + hc + kk * 32 + half * 8;
        qf[kk].q[0] = *reinterpret_cast<const uint4*>(qp);
        qf[kk].q[1] = *reinterpret_cast<const uint4*>(qp + 16);
    }

    const v8f vzero = {0.f, 0.f, 0.f, 0.f, 0.f, 0.f, 0.f, 0.f};
    v8f oacc[4];
#pragma unroll
    for (int nt = 0; nt < 4; ++nt) oacc[nt] = vzero;
    float mrun[8], lrun[8];
#pragma unroll
    for (int j = 0; j < 8; ++j) { mrun[j] = -1e30f; lrun[j] = 0.f; }

    const float sc = 0.03125f;  // 1/sqrt(D) = 1/32 (reference scales by full dim)

    for (int kb = 0; kb < T_; kb += BK) {
        __syncthreads();
        // K tile: async global->LDS (row-major = B-fragment layout for QK^T)
#pragma unroll
        for (int i = 0; i < 2; ++i) {
            int c = tid + i * 256;                 // 512 chunks of 8 bf16
            int r = c >> 3, c8 = (c & 7) * 8;
            async_ld16(Kt + r * KP + c8,
                       Kb + rowbase + (size_t)(kb + r) * D_ + hc + c8);
        }
        // V tile (= Q rows) transposed into Vt[d][t] via VGPRs
#pragma unroll
        for (int i = 0; i < 2; ++i) {
            int c = tid + i * 256;
            int r = c >> 3, c8 = (c & 7) * 8;
            uint4 vd = *reinterpret_cast<const uint4*>(
                Qb + rowbase + (size_t)(kb + r) * D_ + hc + c8);
            const unsigned short* vs = reinterpret_cast<const unsigned short*>(&vd);
#pragma unroll
            for (int e = 0; e < 8; ++e) Vt[(c8 + e) * KP + r] = vs[e];
        }
        if (kb + BK < T_) {  // prefetch next key block (global_prefetch_b8)
            const size_t pg =
                rowbase + (size_t)(kb + BK + (tid >> 2)) * D_ + hc + (tid & 3) * 16;
            __builtin_prefetch(Kb + pg, 0, 1);
            __builtin_prefetch(Qb + pg, 0, 1);
        }
        async_wait0();
        __syncthreads();

        // ---- S = Q K^T : B[k=d][n=t] = K[t][d] -> Kt rows are B fragments ----
        v8f sacc[4];
#pragma unroll
        for (int nt = 0; nt < 4; ++nt) sacc[nt] = vzero;
#pragma unroll
        for (int kk = 0; kk < 2; ++kk) {
#pragma unroll
            for (int nt = 0; nt < 4; ++nt) {
                FragBF bf;
                const unsigned short* kp = Kt + (nt * 16 + lc) * KP + kk * 32 + half * 8;
                bf.q[0] = *reinterpret_cast<const uint4*>(kp);
                bf.q[1] = *reinterpret_cast<const uint4*>(kp + 16);
                sacc[nt] = __builtin_amdgcn_wmma_f32_16x16x32_bf16(
                    false, qf[kk].v, false, bf.v, (short)0, sacc[nt], false, false);
            }
        }

        // ---- online softmax: rows live across 16-lane halves in C layout ----
#pragma unroll
        for (int j = 0; j < 8; ++j) {
            float rm = -1e30f;
#pragma unroll
            for (int nt = 0; nt < 4; ++nt) {
                sacc[nt][j] *= sc;
                rm = fmaxf(rm, sacc[nt][j]);
            }
#pragma unroll
            for (int s = 1; s < 16; s <<= 1) rm = fmaxf(rm, __shfl_xor(rm, s, 32));
            float mnew  = fmaxf(mrun[j], rm);
            float alpha = __expf(mrun[j] - mnew);
            float rs = 0.f;
            float p[4];
#pragma unroll
            for (int nt = 0; nt < 4; ++nt) {
                p[nt] = __expf(sacc[nt][j] - mnew);
                rs += p[nt];
            }
#pragma unroll
            for (int s = 1; s < 16; s <<= 1) rs += __shfl_xor(rs, s, 32);
            lrun[j] = lrun[j] * alpha + rs;
            mrun[j] = mnew;
#pragma unroll
            for (int nt = 0; nt < 4; ++nt) {
                oacc[nt][j] *= alpha;
                Pb[(w * 16 + half * 8 + j) * KP + nt * 16 + lc] = f2bf(p[nt]);
            }
        }

        // ---- O += P V : A from Pb, B from Vt (contiguous-t rows) ----
#pragma unroll
        for (int kk = 0; kk < 2; ++kk) {
            FragBF af;
            const unsigned short* pp = Pb + (w * 16 + lc) * KP + kk * 32 + half * 8;
            af.q[0] = *reinterpret_cast<const uint4*>(pp);
            af.q[1] = *reinterpret_cast<const uint4*>(pp + 16);
#pragma unroll
            for (int nt = 0; nt < 4; ++nt) {
                FragBF vf;
                const unsigned short* vp = Vt + (nt * 16 + lc) * KP + kk * 32 + half * 8;
                vf.q[0] = *reinterpret_cast<const uint4*>(vp);
                vf.q[1] = *reinterpret_cast<const uint4*>(vp + 16);
                oacc[nt] = __builtin_amdgcn_wmma_f32_16x16x32_bf16(
                    false, af.v, false, vf.v, (short)0, oacc[nt], false, false);
            }
        }
    }

    // epilogue: O / l -> out[b][q][h*64 + d] (f32)
#pragma unroll
    for (int nt = 0; nt < 4; ++nt) {
#pragma unroll
        for (int j = 0; j < 8; ++j) {
            int qrow = q0 + half * 8 + j;
            out[rowbase + (size_t)qrow * D_ + hc + nt * 16 + lc] = oacc[nt][j] / lrun[j];
        }
    }
}

// ---------------------------------------------------------------------------
// Host entry. ws layout (ushorts): Qb | Kb | xb | Wqb | Wkb  (~28 MB)
// ---------------------------------------------------------------------------
extern "C" void kernel_launch(void* const* d_in, const int* in_sizes, int n_in,
                              void* d_out, int out_size, void* d_ws, size_t ws_size,
                              hipStream_t stream) {
    (void)in_sizes; (void)n_in; (void)out_size; (void)ws_size;
    const float* x  = (const float*)d_in[0];
    const float* Wq = (const float*)d_in[1];
    const float* bq = (const float*)d_in[2];
    const float* Wk = (const float*)d_in[3];
    const float* bk = (const float*)d_in[4];
    float* out = (float*)d_out;

    unsigned short* Qb  = (unsigned short*)d_ws;
    unsigned short* Kb  = Qb  + (size_t)M_ * D_;
    unsigned short* xb  = Kb  + (size_t)M_ * D_;
    unsigned short* Wqb = xb  + (size_t)M_ * D_;
    unsigned short* Wkb = Wqb + (size_t)D_ * D_;

    // f32 -> bf16 packs
    cvt_f32_bf16<<<(M_ * D_ / 8) / 256, 256, 0, stream>>>(x, xb, M_ * D_ / 8);
    cvt_f32_bf16<<<(D_ * D_ / 8) / 256, 256, 0, stream>>>(Wq, Wqb, D_ * D_ / 8);
    cvt_f32_bf16<<<(D_ * D_ / 8) / 256, 256, 0, stream>>>(Wk, Wkb, D_ * D_ / 8);

    // Q = xWq^T + bq ; K = xWk^T + bk  (bf16 outputs; V=Q reuses Qb)
    dim3 g1(D_ / GN, M_ / GM);
    proj_gemm<<<g1, 256, 0, stream>>>(xb, Wqb, bq, Qb);
    proj_gemm<<<g1, 256, 0, stream>>>(xb, Wkb, bk, Kb);

    // flash attention
    dim3 g2(T_ / (NW * 16), B_ * H_);
    attn_kernel<<<g2, 256, 0, stream>>>(Qb, Kb, out);
}
